// HierarchicalModel_22488448761948
// MI455X (gfx1250) — matrix operand order repaired and checked by reading, not compiled, not run on hardware
//
#include <hip/hip_runtime.h>
#include <cstddef>
#include <cstdint>

// ---------------------------------------------------------------------------
// MI455X (gfx1250). Compute-bound hierarchical transformer: all matmuls on
// v_wmma_f32_16x16x32_bf16 (f32 accum). LDS tiles stored in WMMA fragment
// order -> fragment loads are contiguous ds_load_b128 pairs. f32->bf16
// conversion via native v2bf16 packs (v_cvt_pk_bf16_f32). Staging loops issue
// all global loads before converting so loads stay in flight.
// ---------------------------------------------------------------------------

typedef __attribute__((ext_vector_type(16))) __bf16 v16bf;
typedef __attribute__((ext_vector_type(2)))  __bf16 v2bf;
typedef __attribute__((ext_vector_type(8)))  float  v8f;

#define B_N 2
#define L_N 2048
#define H_N 12
#define MT  (B_N * L_N)       // 4096 token rows

#define BM 64
#define BN 128
#define BK 64

static __device__ __forceinline__ unsigned pack2(float x, float y) {
  v2bf v;
  v[0] = (__bf16)x;
  v[1] = (__bf16)y;
  return __builtin_bit_cast(unsigned, v);   // single v_cvt_pk_bf16_f32
}

static __device__ __forceinline__ v8f wmma_bf16(v16bf a, v16bf b, v8f c) {
  return __builtin_amdgcn_wmma_f32_16x16x32_bf16(false, a, false, b, (short)0, c, false, false);
}

// Fragment-order subtile: 512 bf16 (16x32 A-subtile or 32x16 B-subtile).
// A: lane L = (m&15) + 16*kh owns halfwords e=0..15 with k = kh*16 + e.
// B: lane L = (n&15) + 16*(k>=16) owns halfwords e = k&15.
// Lane data = 32 contiguous bytes at subtile + L*16 halfwords.

// ---------------------------------------------------------------------------
// GEMM: C[M,N] = A[M,K] @ B[K,N]  (f32 memory, bf16 WMMA compute, f32 accum)
// epi: 0 none | 1 C = acc + aux[row*N+col] | 2 C = silu(acc + aux[col])
// Requires M%64==0, N%128==0, K%64==0 (true for every shape used here).
// ---------------------------------------------------------------------------
__global__ __launch_bounds__(256)
void gemm_bf16_kernel(const float* __restrict__ A, const float* __restrict__ Bw,
                      float* __restrict__ C, const float* __restrict__ aux,
                      int M, int N, int K, int epi)
{
  __shared__ __attribute__((aligned(32))) __bf16 AsF[BM * BK];   // 8 KB  (8 subtiles)
  __shared__ __attribute__((aligned(32))) __bf16 BsF[BK * BN];   // 16 KB (16 subtiles)
  const int tid  = threadIdx.x;
  const int lane = tid & 31;
  const int wid  = tid >> 5;
  const int wm   = wid >> 2;   // 0..1 : rows wm*32..+31
  const int wn   = wid & 3;    // 0..3 : cols wn*32..+31
  const int m0 = blockIdx.y * BM;
  const int n0 = blockIdx.x * BN;
  (void)M;

  v8f acc[2][2] = {};

  for (int k0 = 0; k0 < K; k0 += BK) {
    // ---- stage A (64x64 f32): issue 4 float4 loads, then convert+store ----
    float4 va[4];
#pragma unroll
    for (int it = 0; it < 4; ++it) {
      const int f4 = tid + it * 256;
      const int r  = f4 >> 4;            // row 0..63
      const int c  = (f4 & 15) << 2;     // col 0..60 step 4
      va[it] = *(const float4*)(A + (size_t)(m0 + r) * K + k0 + c);
    }
    // ---- stage B (64x128 f32): issue 32 strided loads, then convert+store ----
    float vb[2][16];
#pragma unroll
    for (int it = 0; it < 2; ++it) {
      const int t  = tid + it * 256;     // 0..511
      const int n  = t & 127;
      const int kg = t >> 7;             // 0..3 : k16-group
#pragma unroll
      for (int i = 0; i < 16; ++i)
        vb[it][i] = Bw[(size_t)(k0 + kg * 16 + i) * N + n0 + n];
    }
#pragma unroll
    for (int it = 0; it < 4; ++it) {
      const int f4 = tid + it * 256;
      const int r  = f4 >> 4;
      const int c  = (f4 & 15) << 2;
      const int mt2 = r >> 4, m16 = r & 15;
      const int kt2 = c >> 5, kk = c & 31;
      const int kh = kk >> 4, g = (kk >> 3) & 1, rr = kk & 7;   // rr in {0,4}
      unsigned* dst = (unsigned*)&AsF[(mt2 * 2 + kt2) * 512 + (m16 + 16 * kh) * 16 + g * 8 + rr];
      *(uint2*)dst = make_uint2(pack2(va[it].x, va[it].y), pack2(va[it].z, va[it].w));
    }
#pragma unroll
    for (int it = 0; it < 2; ++it) {
      const int t  = tid + it * 256;
      const int n  = t & 127;
      const int kg = t >> 7;
      unsigned w[8];
#pragma unroll
      for (int i = 0; i < 16; i += 2) w[i >> 1] = pack2(vb[it][i], vb[it][i + 1]);
      unsigned* dst = (unsigned*)&BsF[((kg >> 1) * 8 + (n >> 4)) * 512 + ((n & 15) + 16 * (kg & 1)) * 16];
      *(uint4*)(dst + 0) = make_uint4(w[0], w[1], w[2], w[3]);
      *(uint4*)(dst + 4) = make_uint4(w[4], w[5], w[6], w[7]);
    }
    // prefetch next K-tile while this one is consumed (global_prefetch_b8)
    if (k0 + BK < K) {
      __builtin_prefetch(&A[(size_t)(m0 + (tid & 63)) * K + k0 + BK], 0, 0);
      __builtin_prefetch(&Bw[(size_t)(k0 + BK + (tid >> 2)) * N + n0 + ((tid & 3) << 5)], 0, 0);
    }
    __syncthreads();

#pragma unroll
    for (int kt2 = 0; kt2 < 2; ++kt2) {
      const v16bf a0 = *(const v16bf*)&AsF[((wm * 2 + 0) * 2 + kt2) * 512 + lane * 16];
      const v16bf a1 = *(const v16bf*)&AsF[((wm * 2 + 1) * 2 + kt2) * 512 + lane * 16];
      const v16bf b0 = *(const v16bf*)&BsF[(kt2 * 8 + wn * 2 + 0) * 512 + lane * 16];
      const v16bf b1 = *(const v16bf*)&BsF[(kt2 * 8 + wn * 2 + 1) * 512 + lane * 16];
      acc[0][0] = wmma_bf16(a0, b0, acc[0][0]);
      acc[0][1] = wmma_bf16(a0, b1, acc[0][1]);
      acc[1][0] = wmma_bf16(a1, b0, acc[1][0]);
      acc[1][1] = wmma_bf16(a1, b1, acc[1][1]);
    }
    __syncthreads();
  }

  const int ml = (lane >> 4) << 3;   // C layout: m = r + 8*(lane>=16), n = lane&15
  const int nn = lane & 15;
#pragma unroll
  for (int mt = 0; mt < 2; ++mt)
#pragma unroll
    for (int nt = 0; nt < 2; ++nt)
#pragma unroll
      for (int r = 0; r < 8; ++r) {
        const int row = m0 + wm * 32 + mt * 16 + ml + r;
        const int col = n0 + wn * 32 + nt * 16 + nn;
        float v = acc[mt][nt][r];
        if (epi == 1)      v += aux[(size_t)row * N + col];
        else if (epi == 2) { v += aux[col]; v = v / (1.f + __expf(-v)); }
        C[(size_t)row * N + col] = v;
      }
}

// ---------------------------------------------------------------------------
// Causal flash attention. One wave per (b, head, 32-row q tile).
// Q/K/V/O layout: [B, L, H*HD].  Q & P staged in A-fragment order, V staged in
// B-fragment order, K row-major (its transposed fragment is k-contiguous).
// ---------------------------------------------------------------------------
template <int HD>
__global__ __launch_bounds__(32)
void fa_kernel(const float* __restrict__ Q, const float* __restrict__ Kp,
               const float* __restrict__ V, float* __restrict__ O,
               int Lq, int Dm)
{
  constexpr int NT  = HD / 16;    // output n-tiles
  constexpr int KT  = HD / 32;    // k-steps for Q*K^T
  constexpr int LDK = HD + 16;    // row stride (halfwords), multiple of 16 -> 32B
  __shared__ __attribute__((aligned(32))) __bf16 QsF[2 * KT * 512];
  __shared__ __attribute__((aligned(32))) __bf16 KsR[32 * LDK];
  __shared__ __attribute__((aligned(32))) __bf16 VtF[NT * 512];
  __shared__ __attribute__((aligned(32))) __bf16 sPF[2 * 512];
  __shared__ float sS[32][36];
  __shared__ float corr[32];
  __shared__ float lrow[32];

  const int lane = threadIdx.x;
  const int q0 = blockIdx.x * 32;
  const int bh = blockIdx.y;
  const int b = bh / H_N, h = bh % H_N;
  const float scale = rsqrtf((float)HD);
  const int ml  = (lane >> 4) << 3;
  const int nn  = lane & 15;
  const int qi  = q0 + lane;
  const int col = h * HD;

  // ---- stage Q (32 x HD) into A-fragment order ----
  {
    const int m16 = lane & 15, kh = lane >> 4;
#pragma unroll
    for (int mt = 0; mt < 2; ++mt)
#pragma unroll
      for (int kt2 = 0; kt2 < KT; ++kt2) {
        const size_t rb = (size_t)(b * Lq + q0 + mt * 16 + m16) * Dm + col + kt2 * 32;
        const float4 u0 = *(const float4*)&Q[rb + kh * 16];
        const float4 u1 = *(const float4*)&Q[rb + kh * 16 + 4];
        const float4 u2 = *(const float4*)&Q[rb + kh * 16 + 8];
        const float4 u3 = *(const float4*)&Q[rb + kh * 16 + 12];
        unsigned* dst = (unsigned*)&QsF[(mt * KT + kt2) * 512 + lane * 16];
        *(uint4*)(dst + 0) = make_uint4(pack2(u0.x, u0.y), pack2(u0.z, u0.w),
                                        pack2(u1.x, u1.y), pack2(u1.z, u1.w));
        *(uint4*)(dst + 4) = make_uint4(pack2(u2.x, u2.y), pack2(u2.z, u2.w),
                                        pack2(u3.x, u3.y), pack2(u3.z, u3.w));
      }
  }

  v8f acc[2][NT] = {};
  float m_i = -1e30f, l_i = 0.f;
  const int nkv = (q0 >> 5) + 1;   // causal: kv tiles with kv0 <= q0

  for (int kt = 0; kt < nkv; ++kt) {
    const int kv0 = kt * 32;
    const int rowB = b * Lq + kv0;
    if (kt + 1 < nkv) {
      const size_t rn = (size_t)(rowB + 32 + lane) * Dm + col;
      __builtin_prefetch(&Kp[rn], 0, 0);
      __builtin_prefetch(&V[rn], 0, 0);
    }
    // ---- stage K row-major (lane = key row): chunked load-then-store ----
    {
      const size_t rk = (size_t)(rowB + lane) * Dm + col;
#pragma unroll
      for (int c16 = 0; c16 < HD; c16 += 16) {
        float4 kv4[4];
#pragma unroll
        for (int q4 = 0; q4 < 4; ++q4) kv4[q4] = *(const float4*)&Kp[rk + c16 + q4 * 4];
#pragma unroll
        for (int q4 = 0; q4 < 4; ++q4)
          *(uint2*)&KsR[lane * LDK + c16 + q4 * 4] =
              make_uint2(pack2(kv4[q4].x, kv4[q4].y), pack2(kv4[q4].z, kv4[q4].w));
      }
    }
    // ---- stage V transposed into B-fragment order: (d, k16) tasks ----
    for (int t = lane; t < 2 * HD; t += 32) {
      const int kh = t / HD;
      const int d  = t - kh * HD;
      float xv[16];
#pragma unroll
      for (int i = 0; i < 16; ++i)
        xv[i] = V[(size_t)(rowB + kh * 16 + i) * Dm + col + d];
      unsigned w[8];
#pragma unroll
      for (int i = 0; i < 16; i += 2) w[i >> 1] = pack2(xv[i], xv[i + 1]);
      unsigned* dst = (unsigned*)&VtF[(d >> 4) * 512 + ((d & 15) + 16 * kh) * 16];
      *(uint4*)(dst + 0) = make_uint4(w[0], w[1], w[2], w[3]);
      *(uint4*)(dst + 4) = make_uint4(w[4], w[5], w[6], w[7]);
    }
    __syncthreads();

    // ---- S = Q (32xHD) * K^T (HDx32) ----
    v8f sacc[2][2] = {};
#pragma unroll
    for (int kk = 0; kk < KT; ++kk) {
      const v16bf a0 = *(const v16bf*)&QsF[(0 * KT + kk) * 512 + lane * 16];
      const v16bf a1 = *(const v16bf*)&QsF[(1 * KT + kk) * 512 + lane * 16];
      const v16bf b0 = *(const v16bf*)&KsR[( 0 + nn) * LDK + kk * 32 + (lane >> 4) * 16];
      const v16bf b1 = *(const v16bf*)&KsR[(16 + nn) * LDK + kk * 32 + (lane >> 4) * 16];
      sacc[0][0] = wmma_bf16(a0, b0, sacc[0][0]);
      sacc[0][1] = wmma_bf16(a0, b1, sacc[0][1]);
      sacc[1][0] = wmma_bf16(a1, b0, sacc[1][0]);
      sacc[1][1] = wmma_bf16(a1, b1, sacc[1][1]);
    }
#pragma unroll
    for (int mt = 0; mt < 2; ++mt)
#pragma unroll
      for (int nt = 0; nt < 2; ++nt)
#pragma unroll
        for (int r = 0; r < 8; ++r)
          sS[mt * 16 + ml + r][nt * 16 + nn] = sacc[mt][nt][r];
    __syncthreads();

    // ---- online softmax: lane owns q-row `lane` ----
    float tmax = -1e30f;
#pragma unroll
    for (int j4 = 0; j4 < 32; j4 += 4) {
      const float4 sv = *(const float4*)&sS[lane][j4];
      tmax = fmaxf(tmax, (kv0 + j4 + 0 <= qi) ? sv.x * scale : -1e30f);
      tmax = fmaxf(tmax, (kv0 + j4 + 1 <= qi) ? sv.y * scale : -1e30f);
      tmax = fmaxf(tmax, (kv0 + j4 + 2 <= qi) ? sv.z * scale : -1e30f);
      tmax = fmaxf(tmax, (kv0 + j4 + 3 <= qi) ? sv.w * scale : -1e30f);
    }
    const float mnew = fmaxf(m_i, tmax);
    const float cfac = __expf(m_i - mnew);
    float rs = 0.f;
#pragma unroll
    for (int c = 0; c < 4; ++c) {          // P in A-fragment order, 4x b128
      const int khc = c & 1, g = c >> 1;
      const int jb = khc * 16 + g * 8;
      unsigned w[4];
#pragma unroll
      for (int i = 0; i < 8; i += 2) {
        const int j0 = jb + i, j1 = jb + i + 1;
        const float p0 = (kv0 + j0 <= qi) ? __expf(sS[lane][j0] * scale - mnew) : 0.f;
        const float p1 = (kv0 + j1 <= qi) ? __expf(sS[lane][j1] * scale - mnew) : 0.f;
        rs += p0 + p1;
        w[i >> 1] = pack2(p0, p1);
      }
      unsigned* dst = (unsigned*)&sPF[(lane >> 4) * 512 + ((lane & 15) + 16 * khc) * 16 + g * 8];
      *(uint4*)dst = make_uint4(w[0], w[1], w[2], w[3]);
    }
    l_i = l_i * cfac + rs;
    m_i = mnew;
    corr[lane] = cfac;
    __syncthreads();

    // ---- O = O*corr + P (32x32) * V (32xHD) ----
    const v16bf pa0 = *(const v16bf*)&sPF[0 * 512 + lane * 16];
    const v16bf pa1 = *(const v16bf*)&sPF[1 * 512 + lane * 16];
#pragma unroll
    for (int nt = 0; nt < NT; ++nt) {
      const v16bf bv = *(const v16bf*)&VtF[nt * 512 + lane * 16];
#pragma unroll
      for (int mt = 0; mt < 2; ++mt) {
#pragma unroll
        for (int r = 0; r < 8; ++r) acc[mt][nt][r] *= corr[mt * 16 + ml + r];
        acc[mt][nt] = wmma_bf16(mt == 0 ? pa0 : pa1, bv, acc[mt][nt]);
      }
    }
    __syncthreads();
  }

  lrow[lane] = l_i;
  __syncthreads();
#pragma unroll
  for (int mt = 0; mt < 2; ++mt)
#pragma unroll
    for (int nt = 0; nt < NT; ++nt)
#pragma unroll
      for (int r = 0; r < 8; ++r) {
        const int m = mt * 16 + ml + r;
        O[(size_t)(b * Lq + q0 + m) * Dm + col + nt * 16 + nn] = acc[mt][nt][r] / lrow[m];
      }
}

// ---------------------------------------------------------------------------
// Elementwise / reduction helpers
// ---------------------------------------------------------------------------
__global__ __launch_bounds__(256)
void rmsnorm_kernel(const float* __restrict__ x, const float* __restrict__ w,
                    float* __restrict__ o, int dim)
{
  const int row = blockIdx.x;
  const int tid = threadIdx.x;
  const float* xr = x + (size_t)row * dim;
  float s = 0.f;
  for (int i = tid; i < dim; i += 256) { const float v = xr[i]; s += v * v; }
#pragma unroll
  for (int off = 16; off > 0; off >>= 1) s += __shfl_xor(s, off, 32);
  __shared__ float red[8];
  if ((tid & 31) == 0) red[tid >> 5] = s;
  __syncthreads();
  float tot = 0.f;
#pragma unroll
  for (int i = 0; i < 8; ++i) tot += red[i];
  const float inv = rsqrtf(tot / (float)dim + 1e-5f);
  float* orow = o + (size_t)row * dim;
  for (int i = tid; i < dim; i += 256) orow[i] = xr[i] * inv * w[i];
}

// lt2 = -2*log2(ROPE_BASE)/hd, so theta_j = exp2(j * lt2)
__global__ void rope_kernel(float* __restrict__ x, int Ln, int Dm, int hd,
                            float lt2, int total)
{
  const int i = blockIdx.x * 256 + threadIdx.x;
  if (i >= total) return;
  const int ppr  = Dm >> 1;
  const int row  = i / ppr;
  const int pr   = i % ppr;
  const int half = hd >> 1;
  const int h = pr / half;
  const int j = pr % half;
  const int t = row % Ln;
  const float theta = exp2f((float)j * lt2);
  float sn, cs;
  __sincosf((float)t * theta, &sn, &cs);
  float* base = x + (size_t)row * Dm + h * hd;
  const float x1 = base[j], x2 = base[j + half];
  base[j]        = x1 * cs - x2 * sn;
  base[j + half] = x2 * cs + x1 * sn;
}

__global__ void silu_mul_kernel(float* __restrict__ a, const float* __restrict__ b, size_t n)
{
  const size_t i = (size_t)blockIdx.x * 256 + threadIdx.x;
  if (i >= n) return;
  const float v = a[i];
  a[i] = (v / (1.f + __expf(-v))) * b[i];
}

__global__ __launch_bounds__(256)
void router_head_kernel(const float* __restrict__ r1, const float* __restrict__ w2,
                        const float* __restrict__ b2, float* __restrict__ bmask,
                        int Kd, int Ln)
{
  const int row = blockIdx.x, tid = threadIdx.x;
  float s = 0.f;
  for (int i = tid; i < Kd; i += 256) s += r1[(size_t)row * Kd + i] * w2[i];
#pragma unroll
  for (int off = 16; off > 0; off >>= 1) s += __shfl_xor(s, off, 32);
  __shared__ float red[8];
  if ((tid & 31) == 0) red[tid >> 5] = s;
  __syncthreads();
  if (tid == 0) {
    float tot = 0.f;
#pragma unroll
    for (int i = 0; i < 8; ++i) tot += red[i];
    const float logit = tot + b2[0];
    const float p = 1.f / (1.f + __expf(-logit));
    float hard = (p > 0.5f) ? 1.f : 0.f;
    if ((row % Ln) == 0) hard = 1.f;     // bmask[:,0] = 1
    bmask[row] = hard;
  }
}

// cum[l] = #selected <= l; idx[s] = position of (s+1)-th selected token.
// reference's chunk_ids == cum-1; boundaries[:counts] == idx.
__global__ void router_scan_kernel(const float* __restrict__ bmask, int* __restrict__ cum,
                                   int* __restrict__ idx, int* __restrict__ counts,
                                   int Bn, int Ln)
{
  const int b = threadIdx.x;
  if (b >= Bn) return;
  int c = 0;
  for (int l = 0; l < Ln; ++l) {
    if (bmask[(size_t)b * Ln + l] > 0.5f) { idx[(size_t)b * Ln + c] = l; ++c; }
    cum[(size_t)b * Ln + l] = c;
  }
  counts[b] = c;
}

__global__ void gather_kernel(const float* __restrict__ h, const int* __restrict__ idx,
                              const int* __restrict__ counts, float* __restrict__ out,
                              int Ln, int Dm, size_t total)
{
  const size_t i = (size_t)blockIdx.x * 256 + threadIdx.x;
  if (i >= total) return;
  const int row = (int)(i / Dm);
  const int d   = (int)(i % Dm);
  const int b = row / Ln, s = row % Ln;
  out[i] = (s < counts[b]) ? h[((size_t)b * Ln + idx[row]) * Dm + d] : 0.f;
}

__global__ void combine_kernel(const float* __restrict__ xp, const float* __restrict__ hk,
                               const int* __restrict__ cum, float* __restrict__ y,
                               int Ln, int Dm, size_t total)
{
  const size_t i = (size_t)blockIdx.x * 256 + threadIdx.x;
  if (i >= total) return;
  const int row = (int)(i / Dm);
  const int d   = (int)(i % Dm);
  const int b = row / Ln;
  const int chunk = cum[row] - 1;
  y[i] = xp[((size_t)b * Ln + chunk) * Dm + d] + hk[i];
}

// ---------------------------------------------------------------------------
// Host orchestration (S padded to L: exact for valid rows, static shapes)
// ---------------------------------------------------------------------------
extern "C" void kernel_launch(void* const* d_in, const int* in_sizes, int n_in,
                              void* d_out, int out_size, void* d_ws, size_t ws_size,
                              hipStream_t stream)
{
  (void)in_sizes; (void)n_in; (void)out_size; (void)ws_size;

  const float* X         = (const float*)d_in[0];
  const float* router_w1 = (const float*)d_in[31];
  const float* router_b1 = (const float*)d_in[32];
  const float* router_w2 = (const float*)d_in[33];
  const float* router_b2 = (const float*)d_in[34];
  const float* up_w      = (const float*)d_in[35];
  const float* down_w    = (const float*)d_in[36];

  char* wsp = (char*)d_ws;
  auto alloc = [&](size_t bytes) -> void* {
    void* p = (void*)wsp;
    wsp += (bytes + 255) & ~(size_t)255;
    return p;
  };

  float* xa    = (float*)alloc((size_t)MT * 1152 * sizeof(float));
  float* hn    = (float*)alloc((size_t)MT * 1152 * sizeof(float));
  float* qb    = (float*)alloc((size_t)MT * 1152 * sizeof(float));
  float* kb    = (float*)alloc((size_t)MT * 1152 * sizeof(float));
  float* vb    = (float*)alloc((size_t)MT * 1152 * sizeof(float));
  float* ob    = (float*)alloc((size_t)MT * 1152 * sizeof(float));
  float* t1    = (float*)alloc((size_t)MT * 3072 * sizeof(float));
  float* t2    = (float*)alloc((size_t)MT * 3072 * sizeof(float));
  float* hkeep = (float*)alloc((size_t)MT * 768  * sizeof(float));
  float* bmask = (float*)alloc((size_t)MT * sizeof(float));
  int*   cum    = (int*)alloc((size_t)MT * sizeof(int));
  int*   idxb   = (int*)alloc((size_t)MT * sizeof(int));
  int*   counts = (int*)alloc(64 * sizeof(int));

  auto run_stack = [&](int base, int nlayers, int Dm, int I, int hd) {
    const float* wq  = (const float*)d_in[base + 0];
    const float* wk  = (const float*)d_in[base + 1];
    const float* wv  = (const float*)d_in[base + 2];
    const float* wo  = (const float*)d_in[base + 3];
    const float* w1  = (const float*)d_in[base + 4];
    const float* w3  = (const float*)d_in[base + 5];
    const float* w2  = (const float*)d_in[base + 6];
    const float* ln1 = (const float*)d_in[base + 7];
    const float* ln2 = (const float*)d_in[base + 8];
    const size_t dd = (size_t)Dm * Dm, di = (size_t)Dm * I;
    const dim3 gD(Dm / BN, MT / BM), gI(I / BN, MT / BM);
    const dim3 gfa(L_N / 32, B_N * H_N);
    const int pairs = MT * (Dm / 2);
    const float lt2 = -2.0f * log2f(10000.0f) / (float)hd;
    for (int i = 0; i < nlayers; ++i) {
      rmsnorm_kernel<<<MT, 256, 0, stream>>>(xa, ln1 + (size_t)i * Dm, hn, Dm);
      gemm_bf16_kernel<<<gD, 256, 0, stream>>>(hn, wq + i * dd, qb, nullptr, MT, Dm, Dm, 0);
      gemm_bf16_kernel<<<gD, 256, 0, stream>>>(hn, wk + i * dd, kb, nullptr, MT, Dm, Dm, 0);
      gemm_bf16_kernel<<<gD, 256, 0, stream>>>(hn, wv + i * dd, vb, nullptr, MT, Dm, Dm, 0);
      rope_kernel<<<(pairs + 255) / 256, 256, 0, stream>>>(qb, L_N, Dm, hd, lt2, pairs);
      rope_kernel<<<(pairs + 255) / 256, 256, 0, stream>>>(kb, L_N, Dm, hd, lt2, pairs);
      if (hd == 64) fa_kernel<64><<<gfa, 32, 0, stream>>>(qb, kb, vb, ob, L_N, Dm);
      else          fa_kernel<96><<<gfa, 32, 0, stream>>>(qb, kb, vb, ob, L_N, Dm);
      gemm_bf16_kernel<<<gD, 256, 0, stream>>>(ob, wo + i * dd, xa, xa, MT, Dm, Dm, 1);
      rmsnorm_kernel<<<MT, 256, 0, stream>>>(xa, ln2 + (size_t)i * Dm, hn, Dm);
      gemm_bf16_kernel<<<gI, 256, 0, stream>>>(hn, w1 + i * di, t1, nullptr, MT, I, Dm, 0);
      gemm_bf16_kernel<<<gI, 256, 0, stream>>>(hn, w3 + i * di, t2, nullptr, MT, I, Dm, 0);
      const size_t nel = (size_t)MT * I;
      silu_mul_kernel<<<(nel + 255) / 256, 256, 0, stream>>>(t1, t2, nel);
      gemm_bf16_kernel<<<gD, 256, 0, stream>>>(t1, w2 + i * di, xa, xa, MT, Dm, I, 1);
    }
  };

  // ---- compressor trunk ----
  (void)hipMemcpyAsync(xa, X, (size_t)MT * 768 * sizeof(float), hipMemcpyDeviceToDevice, stream);
  run_stack(1, 3, 768, 2048, 64);
  rmsnorm_kernel<<<MT, 256, 0, stream>>>(xa, (const float*)d_in[10], hkeep, 768); // comp_norm

  // ---- router ----
  {
    const dim3 gR(768 / BN, MT / BM);
    gemm_bf16_kernel<<<gR, 256, 0, stream>>>(hkeep, router_w1, t1, router_b1, MT, 768, 768, 2);
    router_head_kernel<<<MT, 256, 0, stream>>>(t1, router_w2, router_b2, bmask, 768, L_N);
    router_scan_kernel<<<1, 32, 0, stream>>>(bmask, cum, idxb, counts, B_N, L_N);
  }

  // ---- compress (gather) + up-projection ----
  {
    const size_t ng = (size_t)MT * 768;
    gather_kernel<<<(unsigned)((ng + 255) / 256), 256, 0, stream>>>(hkeep, idxb, counts, t2, L_N, 768, ng);
    const dim3 gUp(1152 / BN, MT / BM);
    gemm_bf16_kernel<<<gUp, 256, 0, stream>>>(t2, up_w, xa, nullptr, MT, 1152, 768, 0);
  }

  // ---- processor ----
  run_stack(11, 6, 1152, 3072, 96);
  rmsnorm_kernel<<<MT, 256, 0, stream>>>(xa, (const float*)d_in[20], hn, 1152);   // proc_norm
  {
    const dim3 gDown(768 / BN, MT / BM);
    gemm_bf16_kernel<<<gDown, 256, 0, stream>>>(hn, down_w, t1, nullptr, MT, 768, 1152, 0);
    const size_t ng = (size_t)MT * 768;
    combine_kernel<<<(unsigned)((ng + 255) / 256), 256, 0, stream>>>(t1, hkeep, cum, xa, L_N, 768, ng);
  }

  // ---- decoder + final norm ----
  run_stack(21, 3, 768, 2048, 64);
  rmsnorm_kernel<<<MT, 256, 0, stream>>>(xa, (const float*)d_in[30], (float*)d_out, 768); // dec_norm
}
